// GraphConvBlock_7189775253733
// MI455X (gfx1250) — matrix operand order repaired and checked by reading, not compiled, not run on hardware
//
#include <hip/hip_runtime.h>
#include <hip/hip_bf16.h>
#include <math.h>

typedef _Float16 v16h __attribute__((ext_vector_type(16)));
typedef _Float16 v8h  __attribute__((ext_vector_type(8)));
typedef float    v8f  __attribute__((ext_vector_type(8)));

// Problem constants (match reference)
static constexpr int HH    = 128;   // hidden dim
static constexpr int EDF   = 4;     // edge feature dim
static constexpr int KE    = 3 * HH + EDF;  // 388
static constexpr int KEPAD = 416;   // 13 * 32
static constexpr int KN    = 2 * HH;        // 256 = 8 * 32

__device__ __forceinline__ float gelu_exact(float x) {
    return 0.5f * x * (1.0f + erff(x * 0.70710678118654752440f));
}

// ---------------------------------------------------------------------------
// Weight prep: W (K x 128, row-major [in][out], f32)  ->  WT (128 x Kpad, f16)
// so that a WMMA B-fragment is 16 contiguous halves per lane.
// ---------------------------------------------------------------------------
__global__ void prep_weightsT(const float* __restrict__ W, _Float16* __restrict__ WT,
                              int K, int Kpad) {
    int idx = blockIdx.x * blockDim.x + threadIdx.x;
    if (idx >= HH * Kpad) return;
    int o = idx / Kpad;
    int k = idx - o * Kpad;
    float v = (k < K) ? W[(size_t)k * HH + o] : 0.0f;
    WT[idx] = (_Float16)v;
}

// ---------------------------------------------------------------------------
// Fused edge stage: gather -> MLP1 -> LN -> GELU -> MLP2 -> atomic scatter.
// One workgroup (256 threads = 8 waves) handles 16 edges.
// Wave w computes output columns [16w, 16w+16).
// B-fragments for BOTH GEMMs are preloaded into registers at kernel entry so
// the global loads overlap the gather/LDS-fill stage and the WMMA loops only
// wait on LDS (A-fragment) traffic.
// ---------------------------------------------------------------------------
__global__ __launch_bounds__(256) void edge_stage(
    const float* __restrict__ nf,          // (N, 128)
    const long long* __restrict__ ei,      // (2, E) int64
    const float* __restrict__ ef,          // (E, 4)
    const _Float16* __restrict__ W1T,      // (128, 416) f16
    const float* __restrict__ b1,
    const float* __restrict__ g1, const float* __restrict__ bt1,
    const _Float16* __restrict__ W2T,      // (128, 128) f16
    const float* __restrict__ b2,
    float* __restrict__ agg,               // (N, 128) f32 accum
    float* __restrict__ cnt,               // (N,) f32 accum
    int Etot)
{
    __shared__ __align__(16) _Float16 sA[16][KEPAD];   // edge_in tile, f16
    __shared__ float sH[16][HH];                       // MLP1 output, f32
    __shared__ float sR1[16][16];
    __shared__ float sR2[16][16];
    __shared__ float sMean[16];
    __shared__ float sRstd[16];
    __shared__ __align__(16) _Float16 sA2[16][HH];     // post-GELU, f16
    __shared__ int sSrc[16];
    __shared__ int sDst[16];

    const int tid = threadIdx.x;
    const int e0  = blockIdx.x * 16;

    const int lane  = tid & 31;
    const int wv    = tid >> 5;
    const int col   = (wv << 4) + (lane & 15);
    const int upper = (lane >> 4) & 1;       // lanes 16..31
    const int mbase = upper ? 8 : 0;
    const int arow  = lane & 15;
    const int akoff = upper ? 16 : 0;        // A fragment K base (contiguous, matches B)
    const int bkoff = upper ? 16 : 0;        // B fragment K base

    // ---- preload all B-fragments (weights, L2-hot) -----------------------
    const _Float16* w1p = W1T + (size_t)col * KEPAD + bkoff;
    const _Float16* w2p = W2T + (size_t)col * HH    + bkoff;
    __builtin_prefetch(w1p, 0, 3);           // global_prefetch_b8
    __builtin_prefetch(w2p, 0, 3);
    v16h b1fr[KEPAD / 32];
    #pragma unroll
    for (int k = 0; k < KEPAD / 32; ++k)
        b1fr[k] = *(const v16h*)(w1p + k * 32);
    v16h b2fr[HH / 32];
    #pragma unroll
    for (int k = 0; k < HH / 32; ++k)
        b2fr[k] = *(const v16h*)(w2p + k * 32);

    if (tid < 16) {
        sSrc[tid] = (int)ei[e0 + tid];
        sDst[tid] = (int)ei[(size_t)Etot + e0 + tid];
    }
    __syncthreads();

    // ---- build A tile: [nf_src | nf_dst | ef | nf_src - nf_dst | pad] -----
    {
        const int row = tid >> 4;
        const int t   = tid & 15;
        const int src = sSrc[row];
        const int dst = sDst[row];
        const float* ns = nf + (size_t)src * HH;
        const float* nd = nf + (size_t)dst * HH;
        for (int c = t; c < KEPAD; c += 16) {
            float v;
            if (c < HH)                v = ns[c];
            else if (c < 2 * HH)       v = nd[c - HH];
            else if (c < 2 * HH + EDF) v = ef[(size_t)(e0 + row) * EDF + (c - 2 * HH)];
            else if (c < KE)           { int cc = c - (2 * HH + EDF); v = ns[cc] - nd[cc]; }
            else                       v = 0.0f;
            sA[row][c] = (_Float16)v;
        }
    }
    __syncthreads();

    // ---- MLP1: 16x416 @ 416x16 via 13 WMMAs ------------------------------
    v8f acc = {};
    #pragma unroll
    for (int k = 0; k < KEPAD / 32; ++k) {
        union { v16h v; struct { v8h l; v8h h; } p; } a;
        a.p.l = *(const v8h*)&sA[arow][k * 32 + akoff];
        a.p.h = *(const v8h*)&sA[arow][k * 32 + akoff + 8];
        acc = __builtin_amdgcn_wmma_f32_16x16x32_f16(
            false, a.v, false, b1fr[k], (short)0, acc, false, false);
    }
    {
        const float bias = b1[col];
        #pragma unroll
        for (int r = 0; r < 8; ++r) sH[mbase + r][col] = acc[r] + bias;
    }
    __syncthreads();

    // ---- LayerNorm (row-wise over 128) -----------------------------------
    {
        const int row = tid >> 4;
        const int t   = tid & 15;
        float s1 = 0.0f, s2 = 0.0f;
        #pragma unroll
        for (int j = 0; j < 8; ++j) {
            float x = sH[row][t * 8 + j];
            s1 += x; s2 += x * x;
        }
        sR1[row][t] = s1; sR2[row][t] = s2;
    }
    __syncthreads();
    if (tid < 16) {
        float s1 = 0.0f, s2 = 0.0f;
        #pragma unroll
        for (int j = 0; j < 16; ++j) { s1 += sR1[tid][j]; s2 += sR2[tid][j]; }
        float mu  = s1 * (1.0f / HH);
        float var = s2 * (1.0f / HH) - mu * mu;
        sMean[tid] = mu;
        sRstd[tid] = rsqrtf(var + 1e-5f);
    }
    __syncthreads();
    {
        const int row = tid >> 4;
        const int t   = tid & 15;
        const float mu = sMean[row], rs = sRstd[row];
        #pragma unroll
        for (int j = 0; j < 8; ++j) {
            int c = t * 8 + j;
            float x = (sH[row][c] - mu) * rs * g1[c] + bt1[c];
            sA2[row][c] = (_Float16)gelu_exact(x);
        }
    }
    __syncthreads();

    // ---- MLP2: 16x128 @ 128x16 via 4 WMMAs -------------------------------
    v8f acc2 = {};
    #pragma unroll
    for (int k = 0; k < HH / 32; ++k) {
        union { v16h v; struct { v8h l; v8h h; } p; } a;
        a.p.l = *(const v8h*)&sA2[arow][k * 32 + akoff];
        a.p.h = *(const v8h*)&sA2[arow][k * 32 + akoff + 8];
        acc2 = __builtin_amdgcn_wmma_f32_16x16x32_f16(
            false, a.v, false, b2fr[k], (short)0, acc2, false, false);
    }

    // ---- scatter messages into agg[dst] (f32 atomics, L2-resident) -------
    {
        const float bias = b2[col];
        #pragma unroll
        for (int r = 0; r < 8; ++r) {
            int d = sDst[mbase + r];
            atomicAdd(&agg[(size_t)d * HH + col], acc2[r] + bias);
        }
    }
    if (tid < 16) atomicAdd(&cnt[sDst[tid]], 1.0f);
}

// ---------------------------------------------------------------------------
// Node stage: [nf | agg/count] -> MLP1 -> LN -> GELU -> MLP2 -> + residual.
// One workgroup handles 16 nodes; wave w computes columns [16w, 16w+16).
// ---------------------------------------------------------------------------
__global__ __launch_bounds__(256) void node_stage(
    const float* __restrict__ nf,
    const float* __restrict__ agg,
    const float* __restrict__ cnt,
    const _Float16* __restrict__ W1T,      // (128, 256) f16
    const float* __restrict__ b1,
    const float* __restrict__ g1, const float* __restrict__ bt1,
    const _Float16* __restrict__ W2T,      // (128, 128) f16
    const float* __restrict__ b2,
    float* __restrict__ out)
{
    __shared__ __align__(16) _Float16 sA[16][KN];
    __shared__ float sH[16][HH];
    __shared__ float sR1[16][16];
    __shared__ float sR2[16][16];
    __shared__ float sMean[16];
    __shared__ float sRstd[16];
    __shared__ __align__(16) _Float16 sA2[16][HH];

    const int tid = threadIdx.x;
    const int n0  = blockIdx.x * 16;

    const int lane  = tid & 31;
    const int wv    = tid >> 5;
    const int col   = (wv << 4) + (lane & 15);
    const int upper = (lane >> 4) & 1;
    const int mbase = upper ? 8 : 0;
    const int arow  = lane & 15;
    const int akoff = upper ? 16 : 0;
    const int bkoff = upper ? 16 : 0;

    // ---- preload all B-fragments -----------------------------------------
    const _Float16* w1p = W1T + (size_t)col * KN + bkoff;
    const _Float16* w2p = W2T + (size_t)col * HH + bkoff;
    __builtin_prefetch(w1p, 0, 3);
    __builtin_prefetch(w2p, 0, 3);
    v16h b1fr[KN / 32];
    #pragma unroll
    for (int k = 0; k < KN / 32; ++k)
        b1fr[k] = *(const v16h*)(w1p + k * 32);
    v16h b2fr[HH / 32];
    #pragma unroll
    for (int k = 0; k < HH / 32; ++k)
        b2fr[k] = *(const v16h*)(w2p + k * 32);

    // ---- build A tile: [nf | agg/(count+1e-8)] ---------------------------
    {
        const int row  = tid >> 4;
        const int t    = tid & 15;
        const int node = n0 + row;
        const float invc = 1.0f / (cnt[node] + 1e-8f);
        const float* nrow = nf  + (size_t)node * HH;
        const float* arow_p = agg + (size_t)node * HH;
        for (int c = t; c < KN; c += 16) {
            float v = (c < HH) ? nrow[c] : arow_p[c - HH] * invc;
            sA[row][c] = (_Float16)v;
        }
    }
    __syncthreads();

    // ---- MLP1: 16x256 @ 256x16 via 8 WMMAs -------------------------------
    v8f acc = {};
    #pragma unroll
    for (int k = 0; k < KN / 32; ++k) {
        union { v16h v; struct { v8h l; v8h h; } p; } a;
        a.p.l = *(const v8h*)&sA[arow][k * 32 + akoff];
        a.p.h = *(const v8h*)&sA[arow][k * 32 + akoff + 8];
        acc = __builtin_amdgcn_wmma_f32_16x16x32_f16(
            false, a.v, false, b1fr[k], (short)0, acc, false, false);
    }
    {
        const float bias = b1[col];
        #pragma unroll
        for (int r = 0; r < 8; ++r) sH[mbase + r][col] = acc[r] + bias;
    }
    __syncthreads();

    // ---- LayerNorm -------------------------------------------------------
    {
        const int row = tid >> 4;
        const int t   = tid & 15;
        float s1 = 0.0f, s2 = 0.0f;
        #pragma unroll
        for (int j = 0; j < 8; ++j) {
            float x = sH[row][t * 8 + j];
            s1 += x; s2 += x * x;
        }
        sR1[row][t] = s1; sR2[row][t] = s2;
    }
    __syncthreads();
    if (tid < 16) {
        float s1 = 0.0f, s2 = 0.0f;
        #pragma unroll
        for (int j = 0; j < 16; ++j) { s1 += sR1[tid][j]; s2 += sR2[tid][j]; }
        float mu  = s1 * (1.0f / HH);
        float var = s2 * (1.0f / HH) - mu * mu;
        sMean[tid] = mu;
        sRstd[tid] = rsqrtf(var + 1e-5f);
    }
    __syncthreads();
    {
        const int row = tid >> 4;
        const int t   = tid & 15;
        const float mu = sMean[row], rs = sRstd[row];
        #pragma unroll
        for (int j = 0; j < 8; ++j) {
            int c = t * 8 + j;
            float x = (sH[row][c] - mu) * rs * g1[c] + bt1[c];
            sA2[row][c] = (_Float16)gelu_exact(x);
        }
    }
    __syncthreads();

    // ---- MLP2 + residual -------------------------------------------------
    v8f acc2 = {};
    #pragma unroll
    for (int k = 0; k < HH / 32; ++k) {
        union { v16h v; struct { v8h l; v8h h; } p; } a;
        a.p.l = *(const v8h*)&sA2[arow][k * 32 + akoff];
        a.p.h = *(const v8h*)&sA2[arow][k * 32 + akoff + 8];
        acc2 = __builtin_amdgcn_wmma_f32_16x16x32_f16(
            false, a.v, false, b2fr[k], (short)0, acc2, false, false);
    }
    {
        const float bias = b2[col];
        #pragma unroll
        for (int r = 0; r < 8; ++r) {
            int node = n0 + mbase + r;
            out[(size_t)node * HH + col] =
                acc2[r] + bias + nf[(size_t)node * HH + col];
        }
    }
}

// ---------------------------------------------------------------------------
extern "C" void kernel_launch(void* const* d_in, const int* in_sizes, int n_in,
                              void* d_out, int out_size, void* d_ws, size_t ws_size,
                              hipStream_t stream) {
    const float*     nf   = (const float*)d_in[0];
    const long long* ei   = (const long long*)d_in[1];   // int64 edge_index
    const float*     ef   = (const float*)d_in[2];
    const float*     W1e  = (const float*)d_in[3];
    const float*     b1e  = (const float*)d_in[4];
    const float*     g1e  = (const float*)d_in[5];
    const float*     bt1e = (const float*)d_in[6];
    const float*     W2e  = (const float*)d_in[7];
    const float*     b2e  = (const float*)d_in[8];
    const float*     W1n  = (const float*)d_in[9];
    const float*     b1n  = (const float*)d_in[10];
    const float*     g1n  = (const float*)d_in[11];
    const float*     bt1n = (const float*)d_in[12];
    const float*     W2n  = (const float*)d_in[13];
    const float*     b2n  = (const float*)d_in[14];

    const int Nn = in_sizes[0] / HH;       // 50000
    const int Ee = in_sizes[2] / EDF;      // 800000

    // Workspace layout
    char* ws = (char*)d_ws;
    size_t off = 0;
    _Float16* W1eT = (_Float16*)(ws + off); off += (size_t)HH * KEPAD * sizeof(_Float16);
    _Float16* W2eT = (_Float16*)(ws + off); off += (size_t)HH * HH    * sizeof(_Float16);
    _Float16* W1nT = (_Float16*)(ws + off); off += (size_t)HH * KN    * sizeof(_Float16);
    _Float16* W2nT = (_Float16*)(ws + off); off += (size_t)HH * HH    * sizeof(_Float16);
    off = (off + 255) & ~(size_t)255;
    float* agg = (float*)(ws + off);        off += (size_t)Nn * HH * sizeof(float);
    float* cnt = (float*)(ws + off);        off += (size_t)Nn * sizeof(float);

    // Zero accumulation buffers (agg and cnt are contiguous)
    hipMemsetAsync(agg, 0, (size_t)Nn * (HH + 1) * sizeof(float), stream);

    // Convert + transpose weights to f16 (L2-hot, reused by all blocks)
    prep_weightsT<<<(HH * KEPAD + 255) / 256, 256, 0, stream>>>(W1e, W1eT, KE, KEPAD);
    prep_weightsT<<<(HH * HH    + 255) / 256, 256, 0, stream>>>(W2e, W2eT, HH, HH);
    prep_weightsT<<<(HH * KN    + 255) / 256, 256, 0, stream>>>(W1n, W1nT, KN, KN);
    prep_weightsT<<<(HH * HH    + 255) / 256, 256, 0, stream>>>(W2n, W2nT, HH, HH);

    // Fused edge stage (E/16 workgroups) then node stage (N/16 workgroups)
    edge_stage<<<Ee / 16, 256, 0, stream>>>(nf, ei, ef, W1eT, b1e, g1e, bt1e,
                                            W2eT, b2e, agg, cnt, Ee);
    node_stage<<<Nn / 16, 256, 0, stream>>>(nf, agg, cnt, W1nT, b1n, g1n, bt1n,
                                            W2nT, b2n, (float*)d_out);
}
